// SpatialAttentionModule_12678743458040
// MI455X (gfx1250) — matrix operand order repaired and checked by reading, not compiled, not run on hardware
//
#include <hip/hip_runtime.h>
#include <hip/hip_bf16.h>
#include <math.h>

// MI455X / gfx1250, wave32. bf16 WMMA (V_WMMA_F32_16X16X32_BF16), f32 accum.
// Flash-attention with block-shared, double-buffered K/V tiles in LDS,
// staged via gfx1250 async global->LDS (ASYNCcnt) when available.

typedef __attribute__((ext_vector_type(16))) __bf16 v16bf;
typedef __attribute__((ext_vector_type(8)))  __bf16 v8bf;
typedef __attribute__((ext_vector_type(4)))  __bf16 v4bf;
typedef __attribute__((ext_vector_type(8)))  float  v8f;
typedef __attribute__((ext_vector_type(4)))  int    v4i;

#define NPIX 4096   // H*W
#define CCH  256    // channels
#define BATCH 4
#define JBLK 32     // j-columns staged per iteration

__device__ __forceinline__ v8f zero_v8f() {
  v8f z = {0.f,0.f,0.f,0.f,0.f,0.f,0.f,0.f};
  return z;
}

__device__ __forceinline__ v8f wmma_bf16(v16bf a, v16bf b, v8f c) {
  return __builtin_amdgcn_wmma_f32_16x16x32_bf16(
      /*neg_a=*/false, a, /*neg_b=*/false, b,
      /*c_mod=*/(short)0, c, /*reuse_a=*/false, /*reuse_b=*/false);
}

__device__ __forceinline__ void wait_ds0() {
#if __has_builtin(__builtin_amdgcn_s_wait_dscnt)
  __builtin_amdgcn_s_wait_dscnt(0);
#else
  asm volatile("s_wait_dscnt 0" ::: "memory");
#endif
}

// ---- gfx1250 async global->LDS copy (ASYNCcnt) with safe fallback ----
#if __has_builtin(__builtin_amdgcn_global_load_async_to_lds_b128)
#define HAVE_ASYNC_LDS 1
__device__ __forceinline__ void async_cp16(const __bf16* src, __bf16* dst) {
  __builtin_amdgcn_global_load_async_to_lds_b128(
      (__attribute__((address_space(1))) v4i*)(src),
      (__attribute__((address_space(3))) v4i*)(dst),
      0, 0);
}
__device__ __forceinline__ void wait_async0() {
#if __has_builtin(__builtin_amdgcn_s_wait_asynccnt)
  __builtin_amdgcn_s_wait_asynccnt(0);
#else
  asm volatile("s_wait_asynccnt 0" ::: "memory");
#endif
}
#else
#define HAVE_ASYNC_LDS 0
__device__ __forceinline__ void async_cp16(const __bf16* src, __bf16* dst) {
  *(v8bf*)dst = *(const v8bf*)src;       // global_load_b128 + ds_store_b128
}
__device__ __forceinline__ void wait_async0() {}
#endif

// A-matrix fragment, 16(M) x 32(K) bf16, row-major source with given stride.
// Lane lr holds row M=lr; half lh: VGPR0-3 -> K=lh*8+0..7, VGPR4-7 -> K=16+lh*8+0..7
__device__ __forceinline__ v16bf load_a_frag(const __bf16* base, int stride) {
  union { v16bf v; v8bf h[2]; } f;
  const int lane = threadIdx.x & 31;
  const int lr = lane & 15, lh = lane >> 4;
  const __bf16* p = base + (size_t)lr * stride + lh * 8;
  f.h[0] = *(const v8bf*)(p);
  f.h[1] = *(const v8bf*)(p + 16);
  return f.v;
}

// B-matrix fragment, 32(K) x 16(N) bf16. Element (k,n) at baseT[n*stride + k].
// Lane lr holds column N=lr; half lh: VGPR0-3 -> K=lh*8+0..7, VGPR4-7 -> K=16+lh*8+0..7
__device__ __forceinline__ v16bf load_b_frag(const __bf16* baseT, int stride) {
  union { v16bf v; v8bf h[2]; } f;
  const int lane = threadIdx.x & 31;
  const int lr = lane & 15, lh = lane >> 4;
  const __bf16* p = baseT + (size_t)lr * stride + lh * 8;
  f.h[0] = *(const v8bf*)(p);
  f.h[1] = *(const v8bf*)(p + 16);
  return f.v;
}

// ---------------- conversion kernels ----------------

__global__ void cvt_x_kernel(const float* __restrict__ x, __bf16* __restrict__ xt) {
  int t = blockIdx.x * blockDim.x + threadIdx.x;
  int c = t & (CCH - 1);
  int n = (t >> 8) & (NPIX - 1);
  int b = t >> 20;
  xt[t] = (__bf16)x[((size_t)b * CCH + c) * NPIX + n];
}

__global__ void cvt_w_kernel(const float* __restrict__ Wq, const float* __restrict__ Wk,
                             const float* __restrict__ Wv, __bf16* __restrict__ Wb3) {
  int t = blockIdx.x * blockDim.x + threadIdx.x;
  int which = t / (CCH * CCH);
  int idx   = t % (CCH * CCH);
  const float* src = (which == 0) ? Wq : ((which == 1) ? Wk : Wv);
  Wb3[t] = (__bf16)src[idx];
}

// ---------------- QKV projection ----------------
// Yt[n,o] = sum_c xt[n,c] * W[o,c] + bias[o]
__global__ void __launch_bounds__(128)
proj_kernel(const __bf16* __restrict__ xt, const __bf16* __restrict__ Wb3,
            const float* __restrict__ bq, const float* __restrict__ bk,
            const float* __restrict__ bv,
            __bf16* __restrict__ Qt, __bf16* __restrict__ Kt, __bf16* __restrict__ Vm) {
  const int wid  = threadIdx.x >> 5;
  const int lane = threadIdx.x & 31;
  const int lr = lane & 15, lh = lane >> 4;
  const int b  = blockIdx.y;
  const int n0 = (blockIdx.x * 4 + wid) * 16;

  const __bf16* xb = xt + ((size_t)b * NPIX + n0) * CCH;
  v16bf xa[8];
#pragma unroll
  for (int cc = 0; cc < 8; ++cc) xa[cc] = load_a_frag(xb + cc * 32, CCH);

#pragma unroll
  for (int mtx = 0; mtx < 3; ++mtx) {
    const __bf16* W = Wb3 + (size_t)mtx * CCH * CCH;
    const float* bias = (mtx == 0) ? bq : ((mtx == 1) ? bk : bv);
    for (int ot = 0; ot < 16; ++ot) {
      v8f acc = zero_v8f();
#pragma unroll
      for (int cc = 0; cc < 8; ++cc) {
        v16bf wb = load_b_frag(W + (size_t)(ot * 16) * CCH + cc * 32, CCH);
        acc = wmma_bf16(xa[cc], wb, acc);
      }
      const int oc = ot * 16 + lr;
      const float bb = bias[oc];
#pragma unroll
      for (int v = 0; v < 8; ++v) acc[v] += bb;

      if (mtx < 2) {
        __bf16* dst = ((mtx == 0) ? Qt : Kt) + (size_t)b * NPIX * CCH;
#pragma unroll
        for (int v = 0; v < 8; ++v) {
          int r = n0 + v + lh * 8;
          dst[(size_t)r * CCH + oc] = (__bf16)acc[v];
        }
      } else {
        __bf16* dst = Vm + (size_t)b * CCH * NPIX + (size_t)oc * NPIX + n0 + lh * 8;
        v4bf p0, p1;
        p0[0]=(__bf16)acc[0]; p0[1]=(__bf16)acc[1]; p0[2]=(__bf16)acc[2]; p0[3]=(__bf16)acc[3];
        p1[0]=(__bf16)acc[4]; p1[1]=(__bf16)acc[5]; p1[2]=(__bf16)acc[6]; p1[3]=(__bf16)acc[7];
        *(v4bf*)(dst)     = p0;
        *(v4bf*)(dst + 4) = p1;
      }
    }
  }
}

// ---------------- flash attention ----------------
// 4 waves/block, each owns a 16-row i-tile; block shares staged K/V j-tiles.
__global__ void __launch_bounds__(128)
attn_flash_kernel(const __bf16* __restrict__ Qt, const __bf16* __restrict__ Kt,
                  const __bf16* __restrict__ Vm, const float* __restrict__ x,
                  const float* __restrict__ gamma, float* __restrict__ out) {
  // K tile [JBLK rows j][CCH] and V tile [CCH][JBLK], double buffered = 64KB
  __shared__ __align__(128) __bf16 kbuf[2][JBLK * CCH];
  __shared__ __align__(128) __bf16 vbuf[2][CCH * JBLK];
  __shared__ __align__(64)  __bf16 pstage[4][16 * 32];   // per-wave P staging

  const int tid  = threadIdx.x;
  const int wid  = tid >> 5;
  const int lane = tid & 31;
  const int lr = lane & 15, lh = lane >> 4;
  const int b  = blockIdx.y;
  const int i0 = (blockIdx.x * 4 + wid) * 16;

  const __bf16* Qb = Qt + (size_t)b * NPIX * CCH;
  const __bf16* Kb = Kt + (size_t)b * NPIX * CCH;
  const __bf16* Vb = Vm + (size_t)b * CCH * NPIX;

  // Q fragments for the whole K=256 reduction (8 x 16x32)
  v16bf qa[8];
#pragma unroll
  for (int cc = 0; cc < 8; ++cc)
    qa[cc] = load_a_frag(Qb + (size_t)i0 * CCH + cc * 32, CCH);

  v8f o[16];
#pragma unroll
  for (int t = 0; t < 16; ++t) o[t] = zero_v8f();
  float m[8], l[8];
#pragma unroll
  for (int v = 0; v < 8; ++v) { m[v] = -3.0e38f; l[v] = 0.f; }

  __bf16* ps = pstage[wid];
  const float LOG2E = 1.44269504088896340736f;

  // ---- stage tiles for j0 into buffer `buf` (cooperative, 128 threads) ----
  auto stage = [&](int buf, int j0) {
    // K: 16KB fully contiguous at Kb + j0*CCH   (128 thr x 8 x 16B)
    const __bf16* ks = Kb + (size_t)j0 * CCH;
    __bf16* kd = kbuf[buf];
#pragma unroll
    for (int r = 0; r < 8; ++r) {
      int idx = (tid + r * 128) * 8;
      async_cp16(ks + idx, kd + idx);
    }
    // V: 256 rows (channels) x 64B at Vb + c*NPIX + j0
    __bf16* vd = vbuf[buf];
#pragma unroll
    for (int r = 0; r < 2; ++r) {
      int c = tid + r * 128;
      const __bf16* vs = Vb + (size_t)c * NPIX + j0;
      __bf16* vdd = vd + c * JBLK;
#pragma unroll
      for (int q = 0; q < 4; ++q) async_cp16(vs + q * 8, vdd + q * 8);
    }
  };

  int cur = 0;
  stage(cur, 0);
  wait_async0();
  __syncthreads();

  for (int j0 = 0; j0 < NPIX; j0 += JBLK) {
    if (j0 + JBLK < NPIX) stage(cur ^ 1, j0 + JBLK);   // overlap with compute

    const __bf16* kt = kbuf[cur];
    const __bf16* vt = vbuf[cur];

    // ---- S = Q^T K for 16(i) x 32(j): element (k=c, n=j) at kt[j*CCH + c] ----
    v8f s0 = zero_v8f(), s1 = zero_v8f();
#pragma unroll
    for (int cc = 0; cc < 8; ++cc) {
      v16bf kb0 = load_b_frag(kt + cc * 32, CCH);
      v16bf kb1 = load_b_frag(kt + 16 * CCH + cc * 32, CCH);
      s0 = wmma_bf16(qa[cc], kb0, s0);
      s1 = wmma_bf16(qa[cc], kb1, s1);
    }

    // ---- online softmax (row = v + 8*lh, spread across a 16-lane half) ----
#pragma unroll
    for (int v = 0; v < 8; ++v) {
      float t = fmaxf(s0[v], s1[v]);
      t = fmaxf(t, __shfl_xor(t, 1, 16));
      t = fmaxf(t, __shfl_xor(t, 2, 16));
      t = fmaxf(t, __shfl_xor(t, 4, 16));
      t = fmaxf(t, __shfl_xor(t, 8, 16));
      float mn = fmaxf(m[v], t);
      float scale = exp2f((m[v] - mn) * LOG2E);
      float p0 = exp2f((s0[v] - mn) * LOG2E);
      float p1 = exp2f((s1[v] - mn) * LOG2E);
      s0[v] = p0; s1[v] = p1;
      float rs = p0 + p1;
      rs += __shfl_xor(rs, 1, 16);
      rs += __shfl_xor(rs, 2, 16);
      rs += __shfl_xor(rs, 4, 16);
      rs += __shfl_xor(rs, 8, 16);
      l[v] = l[v] * scale + rs;
      m[v] = mn;
#pragma unroll
      for (int t2 = 0; t2 < 16; ++t2) o[t2][v] *= scale;
    }

    // ---- P (f32 D-layout) -> LDS -> bf16 A-layout ----
#pragma unroll
    for (int v = 0; v < 8; ++v) {
      int r = v + lh * 8;
      ps[r * 32 + lr]      = (__bf16)s0[v];
      ps[r * 32 + 16 + lr] = (__bf16)s1[v];
    }
    wait_ds0();
    v16bf pa = load_a_frag(ps, 32);

    // ---- O += P * V^T: element (k=j, n=c) at vt[c*JBLK + j] ----
#pragma unroll
    for (int ct = 0; ct < 16; ++ct) {
      v16bf vb = load_b_frag(vt + (ct * 16) * JBLK, JBLK);
      o[ct] = wmma_bf16(pa, vb, o[ct]);
    }

    wait_async0();        // next tile fully in LDS
    __syncthreads();      // everyone done with `cur`, next tile visible
    cur ^= 1;
  }

  // ---- epilogue: out = gamma * O/l + x ----
  const float g = gamma[0];
  const float* xb = x   + (size_t)b * CCH * NPIX;
  float*       ob = out + (size_t)b * CCH * NPIX;
#pragma unroll
  for (int ct = 0; ct < 16; ++ct) {
    const int c = ct * 16 + lr;
    const int ibase = i0 + lh * 8;
    const float4* xp = (const float4*)(xb + (size_t)c * NPIX + ibase);
    float4*       op = (float4*)(ob + (size_t)c * NPIX + ibase);
    float4 x0 = xp[0], x1 = xp[1];
    float4 r0, r1;
    r0.x = o[ct][0] / l[0] * g + x0.x;
    r0.y = o[ct][1] / l[1] * g + x0.y;
    r0.z = o[ct][2] / l[2] * g + x0.z;
    r0.w = o[ct][3] / l[3] * g + x0.w;
    r1.x = o[ct][4] / l[4] * g + x1.x;
    r1.y = o[ct][5] / l[5] * g + x1.y;
    r1.z = o[ct][6] / l[6] * g + x1.z;
    r1.w = o[ct][7] / l[7] * g + x1.w;
    op[0] = r0;
    op[1] = r1;
  }
}

// ---------------- launch ----------------

extern "C" void kernel_launch(void* const* d_in, const int* in_sizes, int n_in,
                              void* d_out, int out_size, void* d_ws, size_t ws_size,
                              hipStream_t stream) {
  const float* x     = (const float*)d_in[0];
  const float* Wq    = (const float*)d_in[1];
  const float* bq    = (const float*)d_in[2];
  const float* Wk    = (const float*)d_in[3];
  const float* bk    = (const float*)d_in[4];
  const float* Wv    = (const float*)d_in[5];
  const float* bv    = (const float*)d_in[6];
  const float* gamma = (const float*)d_in[7];
  float* out = (float*)d_out;

  char* ws = (char*)d_ws;
  size_t off = 0;
  const size_t nc2 = (size_t)BATCH * NPIX * CCH * 2;   // 8 MB each
  __bf16* xt  = (__bf16*)(ws + off); off += nc2;
  __bf16* Qt  = (__bf16*)(ws + off); off += nc2;
  __bf16* Kt  = (__bf16*)(ws + off); off += nc2;
  __bf16* Vm  = (__bf16*)(ws + off); off += nc2;
  __bf16* Wb3 = (__bf16*)(ws + off); off += (size_t)3 * CCH * CCH * 2;

  const int totalX = BATCH * NPIX * CCH;               // 4 M
  cvt_x_kernel<<<totalX / 256, 256, 0, stream>>>(x, xt);
  cvt_w_kernel<<<(3 * CCH * CCH) / 256, 256, 0, stream>>>(Wq, Wk, Wv, Wb3);
  proj_kernel<<<dim3(NPIX / 64, BATCH), 128, 0, stream>>>(xt, Wb3, bq, bk, bv, Qt, Kt, Vm);
  attn_flash_kernel<<<dim3(NPIX / 64, BATCH), 128, 0, stream>>>(Qt, Kt, Vm, x, gamma, out);
}
